// MultiHeadAttention_4715874091450
// MI455X (gfx1250) — matrix operand order repaired and checked by reading, not compiled, not run on hardware
//
#include <hip/hip_runtime.h>
#include <hip/hip_bf16.h>

#define B_  2
#define S_  2048
#define D_  1024
#define H_  16
#define HD_ 64

typedef __attribute__((ext_vector_type(16))) __bf16 v16bf;
typedef __attribute__((ext_vector_type(8)))  __bf16 v8bf;
typedef __attribute__((ext_vector_type(4)))  __bf16 v4bf;
typedef __attribute__((ext_vector_type(8)))  float  v8f;

#define WMMA_BF16(a, b, c) \
  __builtin_amdgcn_wmma_f32_16x16x32_bf16(false, (a), false, (b), (short)0, (c), false, false)

// ---------------- CDNA5 async global->LDS copy (ASYNCcnt-tracked) ----------------
#if defined(__has_builtin)
#if __has_builtin(__builtin_amdgcn_global_load_async_to_lds_b128)
#define HAVE_ASYNC_LDS 1
#endif
#endif
#ifndef HAVE_ASYNC_LDS
#define HAVE_ASYNC_LDS 0
#endif

typedef int v4i32 __attribute__((vector_size(16)));
typedef __attribute__((address_space(1))) v4i32 gv4i;   // global int4
typedef __attribute__((address_space(3))) v4i32 lv4i;   // LDS int4

static __device__ __forceinline__ void async_copy16(void* lds_dst, const void* gsrc) {
#if HAVE_ASYNC_LDS
  __builtin_amdgcn_global_load_async_to_lds_b128((gv4i*)gsrc, (lv4i*)lds_dst, 0, 0);
#else
  *(uint4*)lds_dst = *(const uint4*)gsrc;
#endif
}
#if HAVE_ASYNC_LDS
#define ASYNC_WAIT(nstr) asm volatile("s_wait_asynccnt " nstr ::: "memory")
#else
#define ASYNC_WAIT(nstr) asm volatile("" ::: "memory")
#endif

// A fragment (16x32, bf16): lane L holds row M=L%16; elements 0-7 at K=(L/16)*8..,
// elements 8-15 at K=16+(L/16)*8..  -> two contiguous 16B reads from row-major [M][K].
static __device__ __forceinline__ v16bf frag_a(const __bf16* p, int ld, int lane) {
  const int g = lane >> 4, r = lane & 15;
  const __bf16* row = p + (size_t)r * ld + g * 8;
  v8bf lo = *(const v8bf*)(row);
  v8bf hi = *(const v8bf*)(row + 16);
  v16bf f;
#pragma unroll
  for (int i = 0; i < 8; ++i) { f[i] = lo[i]; f[i + 8] = hi[i]; }
  return f;
}

// B fragment (32x16, bf16) from B^T stored row-major [N][K]: lane L holds col N=L%16,
// elements 0-7 at K=(L/16)*8.., elements 8-15 at K=16+(L/16)*8.. -> two contiguous 16B reads.
static __device__ __forceinline__ v16bf frag_bt(const __bf16* p, int ld, int lane) {
  const int g = lane >> 4, n = lane & 15;
  const __bf16* row = p + (size_t)n * ld + g * 8;
  v8bf lo = *(const v8bf*)(row);
  v8bf hi = *(const v8bf*)(row + 16);
  v16bf f;
#pragma unroll
  for (int i = 0; i < 8; ++i) { f[i] = lo[i]; f[i + 8] = hi[i]; }
  return f;
}

// ---------------- f32 -> bf16 conversion (vectorized x4) ----------------
__global__ void cvt_f32_bf16(const float* __restrict__ src, __bf16* __restrict__ dst, int n4) {
  int i = blockIdx.x * blockDim.x + threadIdx.x;
  const int stride = gridDim.x * blockDim.x;
  for (; i < n4; i += stride) {
    const float4 v = ((const float4*)src)[i];
    v4bf o;
    o[0] = (__bf16)v.x; o[1] = (__bf16)v.y; o[2] = (__bf16)v.z; o[3] = (__bf16)v.w;
    ((v4bf*)dst)[i] = o;
  }
}

// ---------------- shared GEMM tile machinery ----------------
// Block: 128 threads = 4 waves (2M x 2N). Block tile 64(M) x 128(N), K-step 32.
// Wave tile 32x64 -> 8 WMMAs per K-step. Double-buffered LDS, async staged.
// Per-thread issue per K-step: A 2 chunks + B 4 chunks = 6 (ASYNCcnt <= 6 wait).
static __device__ __forceinline__ void gemm_issue(
    const __bf16* __restrict__ A, const __bf16* __restrict__ W,
    __bf16* As, __bf16* Bs, int m0, int n0, int k0, int tid) {
#pragma unroll
  for (int i = 0; i < 2; ++i) {                 // A: 64x32 bf16 = 256 x 16B chunks
    const int c = tid + i * 128;
    const int r = c >> 2, o = (c & 3) * 8;
    async_copy16(&As[r * 32 + o], &A[(size_t)(m0 + r) * D_ + k0 + o]);
  }
#pragma unroll
  for (int i = 0; i < 4; ++i) {                 // B: 128x32 bf16 = 512 x 16B chunks
    const int c = tid + i * 128;
    const int r = c >> 2, o = (c & 3) * 8;
    async_copy16(&Bs[r * 32 + o], &W[(size_t)(n0 + r) * D_ + k0 + o]);
  }
}

// ---------------- QKV projection GEMM: [4096 x 1024] * [3072 x 1024]^T ----------------
__global__ void __launch_bounds__(128)
qkv_gemm(const __bf16* __restrict__ X, const __bf16* __restrict__ W,
         const float* __restrict__ bias,
         __bf16* __restrict__ Qo, __bf16* __restrict__ Ko, __bf16* __restrict__ Vto) {
  __shared__ __bf16 As[2][64 * 32];
  __shared__ __bf16 Bs[2][128 * 32];
  const int tid = threadIdx.x, lane = tid & 31, wave = tid >> 5;
  const int wm = wave >> 1, wn = wave & 1;
  const int m0 = blockIdx.x * 64, n0 = blockIdx.y * 128;

  v8f acc[2][4] = {};
  const int NK = D_ / 32;
  gemm_issue(X, W, As[0], Bs[0], m0, n0, 0, tid);
  for (int kt = 0; kt < NK; ++kt) {
    const int cur = kt & 1;
    if (kt + 1 < NK) {
      gemm_issue(X, W, As[cur ^ 1], Bs[cur ^ 1], m0, n0, (kt + 1) * 32, tid);
      ASYNC_WAIT("6");
    } else {
      ASYNC_WAIT("0");
    }
    __syncthreads();
    const __bf16* at = As[cur];
    const __bf16* bt = Bs[cur];
    v16bf a0 = frag_a(&at[(wm * 32 + 0) * 32], 32, lane);
    v16bf a1 = frag_a(&at[(wm * 32 + 16) * 32], 32, lane);
#pragma unroll
    for (int ni = 0; ni < 4; ++ni) {
      v16bf bf = frag_bt(&bt[(wn * 64 + ni * 16) * 32], 32, lane);
      acc[0][ni] = WMMA_BF16(a0, bf, acc[0][ni]);
      acc[1][ni] = WMMA_BF16(a1, bf, acc[1][ni]);
    }
    __syncthreads();
  }
  const int g = lane >> 4, ln = lane & 15;
#pragma unroll
  for (int mi = 0; mi < 2; ++mi) {
#pragma unroll
    for (int ni = 0; ni < 4; ++ni) {
      const int n = n0 + wn * 64 + ni * 16 + ln;
      const float bv = bias[n];
      const int h = n / (3 * HD_);
      const int rem = n % (3 * HD_);
      const int t = rem / HD_, hd = rem % HD_;
      const int mb = m0 + wm * 32 + mi * 16 + 8 * g;
#pragma unroll
      for (int v = 0; v < 8; ++v) {
        const int m = mb + v;
        const int bb = m >> 11;               // / S_
        const int s = m & (S_ - 1);
        const int bh = bb * H_ + h;
        const __bf16 val = (__bf16)(acc[mi][ni][v] + bv);
        if (t == 0)      Qo[((size_t)bh * S_ + s) * HD_ + hd] = val;
        else if (t == 1) Ko[((size_t)bh * S_ + s) * HD_ + hd] = val;
        else             Vto[((size_t)bh * HD_ + hd) * S_ + s] = val;
      }
    }
  }
}

// ---------------- Fused flash-style attention ----------------
// grid (B*H, S/64); block 128 = 4 waves; wave handles 16 query rows.
// K/Vt tiles double-buffered via async LDS loads (8 chunks/thread/tile).
static __device__ __forceinline__ void attn_issue(
    const __bf16* __restrict__ K, const __bf16* __restrict__ Vt,
    __bf16* Ks, __bf16* Vs, int bh, int kt, int tid) {
#pragma unroll
  for (int i = 0; i < 4; ++i) {                 // each tile: 64x64 bf16 = 512 chunks
    const int c = tid + i * 128;
    const int r = c >> 3, o = (c & 7) * 8;
    async_copy16(&Ks[r * 64 + o], &K [((size_t)bh * S_ + kt * 64 + r) * HD_ + o]);
    async_copy16(&Vs[r * 64 + o], &Vt[((size_t)bh * HD_ + r) * S_ + kt * 64 + o]);
  }
}

__global__ void __launch_bounds__(128)
attn_kernel(const __bf16* __restrict__ Q, const __bf16* __restrict__ K,
            const __bf16* __restrict__ Vt, __bf16* __restrict__ O) {
  __shared__ __bf16 Ks[2][64 * 64];    // [key][hd]  = B^T for scores
  __shared__ __bf16 Vs[2][64 * 64];    // [hd][key]  = B^T for PV
  __shared__ __bf16 Ps[4][16 * 64];    // per-wave P tile [qrow][key]
  const int tid = threadIdx.x, lane = tid & 31, wave = tid >> 5;
  const int bh = blockIdx.x;
  const int b = bh >> 4, h = bh & 15;
  const int qt = blockIdx.y;
  const int g = lane >> 4, ln = lane & 15;
  const float LSCALE = 0.2f / 8.0f;    // 0.2 / sqrt(HD)

  // Q fragments live in registers for the whole kernel
  const __bf16* qbase = Q + ((size_t)bh * S_ + qt * 64 + wave * 16) * HD_;
  v16bf qf0 = frag_a(qbase, HD_, lane);
  v16bf qf1 = frag_a(qbase + 32, HD_, lane);

  float mrow[8], lrow[8];
  v8f accO[4] = {};
#pragma unroll
  for (int v = 0; v < 8; ++v) { mrow[v] = -1e30f; lrow[v] = 0.f; }

  const int NT = S_ / 64;
  attn_issue(K, Vt, Ks[0], Vs[0], bh, 0, tid);
  for (int kt = 0; kt < NT; ++kt) {
    const int cur = kt & 1;
    if (kt + 1 < NT) {
      attn_issue(K, Vt, Ks[cur ^ 1], Vs[cur ^ 1], bh, kt + 1, tid);
      ASYNC_WAIT("8");
    } else {
      ASYNC_WAIT("0");
    }
    __syncthreads();
    const __bf16* kst = Ks[cur];
    const __bf16* vst = Vs[cur];

    // scores = Q * K^T  (4 N-subtiles of 16 keys)
    v8f sc[4] = {};
#pragma unroll
    for (int j = 0; j < 4; ++j) {
      v16bf kb0 = frag_bt(&kst[(j * 16) * 64 + 0], 64, lane);
      v16bf kb1 = frag_bt(&kst[(j * 16) * 64 + 32], 64, lane);
      sc[j] = WMMA_BF16(qf0, kb0, sc[j]);
      sc[j] = WMMA_BF16(qf1, kb1, sc[j]);
    }

    // online softmax: row max across 4 subtiles + 16 lanes of the half-wave
    float alpha[8];
#pragma unroll
    for (int v = 0; v < 8; ++v) {
      float mx = fmaxf(fmaxf(sc[0][v], sc[1][v]), fmaxf(sc[2][v], sc[3][v]));
#pragma unroll
      for (int off = 1; off < 16; off <<= 1) mx = fmaxf(mx, __shfl_xor(mx, off, 16));
      mx *= LSCALE;
      const float mnew = fmaxf(mrow[v], mx);
      alpha[v] = __expf(mrow[v] - mnew);
      mrow[v] = mnew;
    }

    // probabilities -> per-wave LDS tile (C-layout -> A-layout transpose)
    float rsum[8];
#pragma unroll
    for (int v = 0; v < 8; ++v) rsum[v] = 0.f;
    __bf16* pw = &Ps[wave][0];
#pragma unroll
    for (int j = 0; j < 4; ++j) {
#pragma unroll
      for (int v = 0; v < 8; ++v) {
        const float p = __expf(sc[j][v] * LSCALE - mrow[v]);
        rsum[v] += p;
        pw[(v + 8 * g) * 64 + j * 16 + ln] = (__bf16)p;
      }
    }
#pragma unroll
    for (int v = 0; v < 8; ++v) {
      float rs = rsum[v];
#pragma unroll
      for (int off = 1; off < 16; off <<= 1) rs += __shfl_xor(rs, off, 16);
      lrow[v] = lrow[v] * alpha[v] + rs;
    }
    asm volatile("s_wait_dscnt 0" ::: "memory");  // own-wave LDS RAW

    v16bf pf0 = frag_a(pw, 64, lane);
    v16bf pf1 = frag_a(pw + 32, 64, lane);
#pragma unroll
    for (int j = 0; j < 4; ++j) {
#pragma unroll
      for (int v = 0; v < 8; ++v) accO[j][v] *= alpha[v];
      v16bf vb0 = frag_bt(&vst[(j * 16) * 64 + 0], 64, lane);
      v16bf vb1 = frag_bt(&vst[(j * 16) * 64 + 32], 64, lane);
      accO[j] = WMMA_BF16(pf0, vb0, accO[j]);
      accO[j] = WMMA_BF16(pf1, vb1, accO[j]);
    }
    __syncthreads();
  }

  // normalize and write O as bf16 [B*S, D] (heads interleaved)
#pragma unroll
  for (int v = 0; v < 8; ++v) lrow[v] = 1.0f / lrow[v];
#pragma unroll
  for (int j = 0; j < 4; ++j) {
#pragma unroll
    for (int v = 0; v < 8; ++v) {
      const int s = qt * 64 + wave * 16 + v + 8 * g;
      const int col = h * 64 + j * 16 + ln;
      O[((size_t)b * S_ + s) * D_ + col] = (__bf16)(accO[j][v] * lrow[v]);
    }
  }
}

// ---------------- Output projection GEMM: [4096 x 1024] * [1024 x 1024]^T + bias ----------------
__global__ void __launch_bounds__(128)
out_gemm(const __bf16* __restrict__ A, const __bf16* __restrict__ W,
         const float* __restrict__ bias, float* __restrict__ out) {
  __shared__ __bf16 As[2][64 * 32];
  __shared__ __bf16 Bs[2][128 * 32];
  const int tid = threadIdx.x, lane = tid & 31, wave = tid >> 5;
  const int wm = wave >> 1, wn = wave & 1;
  const int m0 = blockIdx.x * 64, n0 = blockIdx.y * 128;

  v8f acc[2][4] = {};
  const int NK = D_ / 32;
  gemm_issue(A, W, As[0], Bs[0], m0, n0, 0, tid);
  for (int kt = 0; kt < NK; ++kt) {
    const int cur = kt & 1;
    if (kt + 1 < NK) {
      gemm_issue(A, W, As[cur ^ 1], Bs[cur ^ 1], m0, n0, (kt + 1) * 32, tid);
      ASYNC_WAIT("6");
    } else {
      ASYNC_WAIT("0");
    }
    __syncthreads();
    const __bf16* at = As[cur];
    const __bf16* bt = Bs[cur];
    v16bf a0 = frag_a(&at[(wm * 32 + 0) * 32], 32, lane);
    v16bf a1 = frag_a(&at[(wm * 32 + 16) * 32], 32, lane);
#pragma unroll
    for (int ni = 0; ni < 4; ++ni) {
      v16bf bf = frag_bt(&bt[(wn * 64 + ni * 16) * 32], 32, lane);
      acc[0][ni] = WMMA_BF16(a0, bf, acc[0][ni]);
      acc[1][ni] = WMMA_BF16(a1, bf, acc[1][ni]);
    }
    __syncthreads();
  }
  const int g = lane >> 4, ln = lane & 15;
#pragma unroll
  for (int mi = 0; mi < 2; ++mi) {
#pragma unroll
    for (int ni = 0; ni < 4; ++ni) {
      const int n = n0 + wn * 64 + ni * 16 + ln;
      const float bv = bias[n];
      const int mb = m0 + wm * 32 + mi * 16 + 8 * g;
#pragma unroll
      for (int v = 0; v < 8; ++v)
        out[(size_t)(mb + v) * D_ + n] = acc[mi][ni][v] + bv;
    }
  }
}

extern "C" void kernel_launch(void* const* d_in, const int* in_sizes, int n_in,
                              void* d_out, int out_size, void* d_ws, size_t ws_size,
                              hipStream_t stream) {
  const float* x     = (const float*)d_in[0];
  const float* qkv_w = (const float*)d_in[1];
  const float* qkv_b = (const float*)d_in[2];
  const float* out_w = (const float*)d_in[3];
  const float* out_b = (const float*)d_in[4];
  float* out = (float*)d_out;
  char* ws = (char*)d_ws;

  // bf16 workspace layout (bytes)
  __bf16* Xb   = (__bf16*)(ws + 0);                // 8 MiB  [B*S, D]
  __bf16* Wqkv = (__bf16*)(ws + 8388608);          // 6 MiB  [3D, D]
  __bf16* Wout = (__bf16*)(ws + 14680064);         // 2 MiB  [D, D]
  __bf16* Qb   = (__bf16*)(ws + 16777216);         // 8 MiB  [B,H,S,HD]
  __bf16* Kb   = (__bf16*)(ws + 25165824);         // 8 MiB  [B,H,S,HD]
  __bf16* Vtb  = (__bf16*)(ws + 33554432);         // 8 MiB  [B,H,HD,S]
  __bf16* Ob   = (__bf16*)(ws + 41943040);         // 8 MiB  [B*S, D]

  cvt_f32_bf16<<<1024, 256, 0, stream>>>(x,     Xb,   (B_ * S_ * D_) / 4);
  cvt_f32_bf16<<<1024, 256, 0, stream>>>(qkv_w, Wqkv, (3 * D_ * D_) / 4);
  cvt_f32_bf16<<<256,  256, 0, stream>>>(out_w, Wout, (D_ * D_) / 4);

  qkv_gemm<<<dim3(64, 24), 128, 0, stream>>>(Xb, Wqkv, qkv_b, Qb, Kb, Vtb);
  attn_kernel<<<dim3(B_ * H_, S_ / 64), 128, 0, stream>>>(Qb, Kb, Vtb, Ob);
  out_gemm<<<dim3(64, 8), 128, 0, stream>>>(Ob, Wout, out_b, out);
}
